// CustomCellModel_36438502539857
// MI455X (gfx1250) — matrix operand order repaired and checked by reading, not compiled, not run on hardware
//
#include <hip/hip_runtime.h>
#include <hip/hip_bf16.h>
#include <math.h>

// Problem constants (from reference): B=64, T=512, V=32000, E=256, H=512
#define BDIM 64
#define TDIM 512
#define EDIM 256
#define HDIM 512
#define HPAD 520  // LDS row padding (bf16 elems): stride 1040B=260 dwords -> rows 4 banks apart

typedef __attribute__((ext_vector_type(16))) __bf16 v16bf;
typedef __attribute__((ext_vector_type(8)))  __bf16 v8bf;
typedef __attribute__((ext_vector_type(8)))  float  v8f;

__device__ __forceinline__ v8f wmma_bf16(v16bf a, v16bf b, v8f c) {
  // D(f32 16x16) = A(bf16 16x32) * B(bf16 32x16) + C
  return __builtin_amdgcn_wmma_f32_16x16x32_bf16(
      /*neg_a=*/false, a, /*neg_b=*/false, b,
      /*c_mod=*/(short)0, c, /*reuse_a=*/false, /*reuse_b=*/false);
}

// A-fragment assembly from two contiguous 8-half runs (contiguous k order, same as the B loads):
// element e maps to K = K0 + e, K0 = kchunk + 16*(lane>>4)
__device__ __forceinline__ v16bf join8(v8bf lo, v8bf hi) {
  return __builtin_shufflevector(lo, hi, 0,1,2,3,4,5,6,7,8,9,10,11,12,13,14,15);
}

__device__ __forceinline__ v16bf cvt16(float4 r0, float4 r1, float4 r2, float4 r3) {
  v16bf a;
  a[0]=(__bf16)r0.x;  a[1]=(__bf16)r0.y;  a[2]=(__bf16)r0.z;  a[3]=(__bf16)r0.w;
  a[4]=(__bf16)r1.x;  a[5]=(__bf16)r1.y;  a[6]=(__bf16)r1.z;  a[7]=(__bf16)r1.w;
  a[8]=(__bf16)r2.x;  a[9]=(__bf16)r2.y;  a[10]=(__bf16)r2.z; a[11]=(__bf16)r2.w;
  a[12]=(__bf16)r3.x; a[13]=(__bf16)r3.y; a[14]=(__bf16)r3.z; a[15]=(__bf16)r3.w;
  return a;
}

// ---------------------------------------------------------------------------
// Kernel 1: convert + transpose weights to bf16 [N,K] for WMMA B-operand loads.
//   WxT[h*E + e] = Wx[e*H + h]   (E*H elems)
//   WhT[n*H + k] = Wh[k*H + n]   (H*H elems)
// ---------------------------------------------------------------------------
__global__ void rnn_prep_weights(const float* __restrict__ Wx0, const float* __restrict__ Wh0,
                                 const float* __restrict__ Wx1, const float* __restrict__ Wh1,
                                 __bf16* __restrict__ WxT0, __bf16* __restrict__ WxT1,
                                 __bf16* __restrict__ WhT0, __bf16* __restrict__ WhT1) {
  int idx = blockIdx.x * 256 + threadIdx.x;
  int which = blockIdx.y;
  if (which == 0) {
    if (idx < EDIM * HDIM) { int h = idx >> 8, e = idx & 255; WxT0[idx] = (__bf16)Wx0[e * HDIM + h]; }
  } else if (which == 1) {
    if (idx < EDIM * HDIM) { int h = idx >> 8, e = idx & 255; WxT1[idx] = (__bf16)Wx1[e * HDIM + h]; }
  } else if (which == 2) {
    if (idx < HDIM * HDIM) { int n = idx >> 9, k = idx & 511; WhT0[idx] = (__bf16)Wh0[k * HDIM + n]; }
  } else {
    if (idx < HDIM * HDIM) { int n = idx >> 9, k = idx & 511; WhT1[idx] = (__bf16)Wh1[k * HDIM + n]; }
  }
}

// ---------------------------------------------------------------------------
// Kernel 2: xe = emb_table[x] @ Wx + b (both cells), output bf16 in WMMA
// C-FRAGMENT layout. Rows ordered g = t*B + b so one GEMM tile == one
// recurrence C tile. Software pipeline: chunk k+1's A-gather floats and all
// four B panels are issued before chunk k's WMMAs execute, so global latency
// overlaps matrix math instead of serializing on s_wait_loadcnt 0.
// One wave computes a 16x64 strip (A reused x4).
// Grid: 4096 blocks x 256 threads = 32768 strips = 2 cells * 2048 Mtiles * 8 Nstrips.
// ---------------------------------------------------------------------------
__global__ void rnn_xe_gemm(const int* __restrict__ xs, const float* __restrict__ emb,
                            const __bf16* __restrict__ WxT0, const __bf16* __restrict__ WxT1,
                            const float* __restrict__ b0, const float* __restrict__ b1,
                            __bf16* __restrict__ xe0, __bf16* __restrict__ xe1) {
  const int wave = threadIdx.x >> 5;
  const int lane = threadIdx.x & 31;
  const int hi = lane >> 4;
  const int lm = lane & 15;

  const int s    = blockIdx.x * 8 + wave;      // strip id
  const int cell = s >> 14;                    // 16384 strips per cell
  const int rem  = s & 16383;
  const int m0   = (rem >> 3) << 4;            // row tile base (rows g = t*B + b)
  const int n0   = (rem & 7) << 6;             // 64-column strip base

  const __bf16* __restrict__ WxT = cell ? WxT1 : WxT0;
  const float*  __restrict__ bias = cell ? b1 : b0;
  __bf16* __restrict__ xe = cell ? xe1 : xe0;

  // A row for this lane: g = m0 + lm -> b = g&63, t = g>>6; token = x[b*T + t]
  const int g   = m0 + lm;
  const int tok = xs[(g & 63) * TDIM + (g >> 6)];
  const float* __restrict__ arow = emb + (size_t)tok * EDIM;

  const __bf16* __restrict__ brow0 = WxT + (size_t)(n0 +  0 + lm) * EDIM + 16 * hi;
  const __bf16* __restrict__ brow1 = WxT + (size_t)(n0 + 16 + lm) * EDIM + 16 * hi;
  const __bf16* __restrict__ brow2 = WxT + (size_t)(n0 + 32 + lm) * EDIM + 16 * hi;
  const __bf16* __restrict__ brow3 = WxT + (size_t)(n0 + 48 + lm) * EDIM + 16 * hi;

  v8f acc0 = {}, acc1 = {}, acc2 = {}, acc3 = {};

  // ---- pipeline prologue: chunk 0 loads ----
  const int A0 = 16 * hi;
  float4 r0 = *(const float4*)(arow + A0);
  float4 r1 = *(const float4*)(arow + A0 + 4);
  float4 r2 = *(const float4*)(arow + A0 + 8);
  float4 r3 = *(const float4*)(arow + A0 + 12);
  v16bf bv0 = *(const v16bf*)(brow0);
  v16bf bv1 = *(const v16bf*)(brow1);
  v16bf bv2 = *(const v16bf*)(brow2);
  v16bf bv3 = *(const v16bf*)(brow3);

#pragma unroll
  for (int kk = 0; kk < EDIM; kk += 32) {
    const v16bf a = cvt16(r0, r1, r2, r3);
    v16bf nb0 = bv0, nb1 = bv1, nb2 = bv2, nb3 = bv3;  // consume current
    if (kk + 32 < EDIM) {
      // issue next chunk's loads before the WMMAs
      const int An = kk + 32 + 16 * hi;
      r0 = *(const float4*)(arow + An);
      r1 = *(const float4*)(arow + An + 4);
      r2 = *(const float4*)(arow + An + 8);
      r3 = *(const float4*)(arow + An + 12);
      bv0 = *(const v16bf*)(brow0 + kk + 32);
      bv1 = *(const v16bf*)(brow1 + kk + 32);
      bv2 = *(const v16bf*)(brow2 + kk + 32);
      bv3 = *(const v16bf*)(brow3 + kk + 32);
    }
    acc0 = wmma_bf16(a, nb0, acc0);
    acc1 = wmma_bf16(a, nb1, acc1);
    acc2 = wmma_bf16(a, nb2, acc2);
    acc3 = wmma_bf16(a, nb3, acc3);
  }

  // Epilogue: +bias, pack lane's 8 C values into one 16B fragment store.
  const int t  = m0 >> 6;
  const int mt = (m0 >> 4) & 3;
#define XE_EPI(J, ACC)                                                           \
  {                                                                              \
    const int n = n0 + (J) * 16 + lm;                                            \
    const float bvs = bias[n];                                                   \
    v8bf fr;                                                                     \
    _Pragma("unroll") for (int r = 0; r < 8; ++r) fr[r] = (__bf16)(ACC[r] + bvs);\
    *(v8bf*)(xe + (((size_t)t * 4 + mt) * 32 + (n >> 4)) * 256 + lane * 8) = fr; \
  }
  XE_EPI(0, acc0) XE_EPI(1, acc1) XE_EPI(2, acc2) XE_EPI(3, acc3)
#undef XE_EPI
}

// ---------------------------------------------------------------------------
// Kernel 3: sequential recurrence. One 1024-thread workgroup per cell.
// h (64x512 bf16) lives in padded LDS; each step: h = act(xe_t + h @ Wh).
// Wave w owns a 16-column slice: 4 M-tiles x 16 K-chunks = 64 WMMAs per step.
// B panels are software-pipelined from L2; next step's xe fragments prefetched.
// ---------------------------------------------------------------------------
__global__ void rnn_recurrence(const __bf16* __restrict__ xef0, const __bf16* __restrict__ xef1,
                               const __bf16* __restrict__ WhT0, const __bf16* __restrict__ WhT1,
                               float* __restrict__ hf) {
  extern __shared__ __bf16 hbuf[];  // [BDIM][HPAD]

  const int cell = blockIdx.x;
  const __bf16* __restrict__ xef = cell ? xef1 : xef0;
  const __bf16* __restrict__ WhT = cell ? WhT1 : WhT0;
  float* __restrict__ hout = hf + (size_t)cell * BDIM * HDIM;

  const int tid = threadIdx.x;
  const int wave = tid >> 5, lane = tid & 31;
  const int hi = lane >> 4, lm = lane & 15;
  const int n = wave * 16 + lm;  // this wave's output column (C/B layout: N = lane&15)

  for (int i = tid; i < BDIM * HPAD; i += 1024) hbuf[i] = (__bf16)0.0f;
  __syncthreads();

  const __bf16* __restrict__ brow = WhT + (size_t)n * HDIM + 16 * hi;

  for (int t = 0; t < TDIM; ++t) {
    // C init: one 16B fragment load per M-tile (mt stride = 32 tiles * 256 elems).
    const __bf16* __restrict__ ft = xef + ((size_t)t * 128 + wave) * 256 + lane * 8;
    const v8bf f0 = *(const v8bf*)(ft + 0 * 8192);
    const v8bf f1 = *(const v8bf*)(ft + 1 * 8192);
    const v8bf f2 = *(const v8bf*)(ft + 2 * 8192);
    const v8bf f3 = *(const v8bf*)(ft + 3 * 8192);
    v8f acc0, acc1, acc2, acc3;
#pragma unroll
    for (int r = 0; r < 8; ++r) {
      acc0[r] = (float)f0[r]; acc1[r] = (float)f1[r];
      acc2[r] = (float)f2[r]; acc3[r] = (float)f3[r];
    }

    // Prefetch next timestep's fragments into cache while we compute.
    if (t + 1 < TDIM) {
      const __bf16* fn = xef + ((size_t)(t + 1) * 128 + wave) * 256 + lane * 8;
      __builtin_prefetch(fn + 0 * 8192, 0, 1);
      __builtin_prefetch(fn + 1 * 8192, 0, 1);
      __builtin_prefetch(fn + 2 * 8192, 0, 1);
      __builtin_prefetch(fn + 3 * 8192, 0, 1);
    }

    // Software-pipelined B panel: load chunk kk+32 while WMMAs consume chunk kk.
    v16bf bv = *(const v16bf*)(brow);
#pragma unroll 4
    for (int kk = 0; kk < HDIM; kk += 32) {
      const v16bf bcur = bv;
      if (kk + 32 < HDIM) bv = *(const v16bf*)(brow + kk + 32);
      const int K0 = kk + 16 * hi;
      v16bf a;
      a = join8(*(const v8bf*)(hbuf + ( 0 + lm) * HPAD + K0),
                *(const v8bf*)(hbuf + ( 0 + lm) * HPAD + K0 + 8));
      acc0 = wmma_bf16(a, bcur, acc0);
      a = join8(*(const v8bf*)(hbuf + (16 + lm) * HPAD + K0),
                *(const v8bf*)(hbuf + (16 + lm) * HPAD + K0 + 8));
      acc1 = wmma_bf16(a, bcur, acc1);
      a = join8(*(const v8bf*)(hbuf + (32 + lm) * HPAD + K0),
                *(const v8bf*)(hbuf + (32 + lm) * HPAD + K0 + 8));
      acc2 = wmma_bf16(a, bcur, acc2);
      a = join8(*(const v8bf*)(hbuf + (48 + lm) * HPAD + K0),
                *(const v8bf*)(hbuf + (48 + lm) * HPAD + K0 + 8));
      acc3 = wmma_bf16(a, bcur, acc3);
    }

    __syncthreads();  // all reads of h done before overwrite

#define H_WRITE(J, ACC)                                                         \
  {                                                                             \
    _Pragma("unroll") for (int r = 0; r < 8; ++r) {                             \
      const int row = (J) * 16 + r + 8 * hi;                                    \
      float v = ACC[r];                                                         \
      v = (cell == 0) ? tanhf(v) : fmaxf(v, 0.0f);                              \
      hbuf[row * HPAD + n] = (__bf16)v;                                         \
    }                                                                           \
  }
    H_WRITE(0, acc0) H_WRITE(1, acc1) H_WRITE(2, acc2) H_WRITE(3, acc3)
#undef H_WRITE

    __syncthreads();  // h fully updated before next step's reads
  }

  // Export final hidden state (f32) for the head kernel.
  for (int i = tid; i < BDIM * HDIM; i += 1024) {
    const int row = i >> 9, col = i & 511;
    hout[i] = (float)hbuf[row * HPAD + col];
  }
}

// ---------------------------------------------------------------------------
// Kernel 4: out[b] = [h_tanh | h_relu] @ fc_w + fc_b   (64 outputs)
// ---------------------------------------------------------------------------
__global__ void rnn_fc_out(const float* __restrict__ hf, const float* __restrict__ fcw,
                           const float* __restrict__ fcb, float* __restrict__ out) {
  const int b = threadIdx.x;
  if (b < BDIM) {
    const float* h0 = hf + (size_t)b * HDIM;
    const float* h1 = hf + (size_t)BDIM * HDIM + (size_t)b * HDIM;
    float acc = fcb[0];
    for (int j = 0; j < HDIM; ++j) acc += h0[j] * fcw[j] + h1[j] * fcw[HDIM + j];
    out[b] = acc;
  }
}

extern "C" void kernel_launch(void* const* d_in, const int* in_sizes, int n_in,
                              void* d_out, int out_size, void* d_ws, size_t ws_size,
                              hipStream_t stream) {
  const int*   x    = (const int*)d_in[0];
  const float* emb  = (const float*)d_in[1];
  const float* Wx0  = (const float*)d_in[2];
  const float* Wh0  = (const float*)d_in[3];
  const float* b0   = (const float*)d_in[4];
  const float* Wx1  = (const float*)d_in[5];
  const float* Wh1  = (const float*)d_in[6];
  const float* b1   = (const float*)d_in[7];
  const float* fcw  = (const float*)d_in[8];
  const float* fcb  = (const float*)d_in[9];
  float* out = (float*)d_out;

  // Workspace layout (bf16 elems unless noted): ~68.5 MB total
  __bf16* ws   = (__bf16*)d_ws;
  __bf16* WxT0 = ws;                                   // E*H   = 131072
  __bf16* WxT1 = WxT0 + (size_t)EDIM * HDIM;           // 131072
  __bf16* WhT0 = WxT1 + (size_t)EDIM * HDIM;           // H*H   = 262144
  __bf16* WhT1 = WhT0 + (size_t)HDIM * HDIM;           // 262144
  __bf16* xe0  = WhT1 + (size_t)HDIM * HDIM;           // T*B*H = 16777216 (fragment layout)
  __bf16* xe1  = xe0 + (size_t)TDIM * BDIM * HDIM;     // 16777216
  float*  hf   = (float*)(xe1 + (size_t)TDIM * BDIM * HDIM);  // 2*B*H f32

  rnn_prep_weights<<<dim3((HDIM * HDIM) / 256, 4), 256, 0, stream>>>(
      Wx0, Wh0, Wx1, Wh1, WxT0, WxT1, WhT0, WhT1);

  rnn_xe_gemm<<<4096, 256, 0, stream>>>(x, emb, WxT0, WxT1, b0, b1, xe0, xe1);

  rnn_recurrence<<<2, 1024, (size_t)BDIM * HPAD * sizeof(__bf16), stream>>>(
      xe0, xe1, WhT0, WhT1, hf);

  rnn_fc_out<<<1, 64, 0, stream>>>(hf, fcw, fcb, out);
}
